// MultiScaleRetention_25847113187577
// MI455X (gfx1250) — matrix operand order repaired and checked by reading, not compiled, not run on hardware
//
#include <hip/hip_runtime.h>
#include <hip/hip_bf16.h>

// ---------------------------------------------------------------------------
// MultiScaleRetention forward for MI455X (gfx1250, wave32, WMMA).
// All GEMMs use v_wmma_f32_16x16x32_bf16 (bf16 in, f32 accumulate).
// ---------------------------------------------------------------------------

typedef __attribute__((ext_vector_type(16))) __bf16 v16bf;
typedef __attribute__((ext_vector_type(8)))  float  v8f;

#define B_  2
#define N_  2048
#define H_  16
#define D_  128
#define E_  2048
#define C_  64
#define NC_ (N_ / C_)
#define M_  (B_ * N_)     // 4096 rows for all GEMMs

static __device__ __forceinline__ unsigned short f2bf(float f) {
  unsigned int u = __float_as_uint(f);
  unsigned int r = 0x7FFFu + ((u >> 16) & 1u);   // round-to-nearest-even
  return (unsigned short)((u + r) >> 16);
}
static __device__ __forceinline__ float bf2f(unsigned short h) {
  return __uint_as_float(((unsigned int)h) << 16);
}

union FragU { uint4 u[2]; v16bf v; };

// A fragment: 16x32 bf16, row-major source with leading dim ld (elements).
// CDNA5 layout: lane = row(0..15) | khalf; lanes<16 hold K {0..7,16..23},
// lanes>=16 hold K {8..15,24..31}  -> two 16B loads per lane.
static __device__ __forceinline__ v16bf load_a(const unsigned short* base, int ld, int lane) {
  int row = lane & 15, kh = lane >> 4;
  const unsigned short* p = base + (size_t)row * (size_t)ld + kh * 8;
  FragU f;
  f.u[0] = *(const uint4*)(p);
  f.u[1] = *(const uint4*)(p + 16);
  return f.v;
}
// B fragment: 32x16 bf16; source laid out [n][k] (column n of B contiguous
// over K with leading dim ld). lane = col(0..15) | khalf; K {0..7,16..23} / {8..15,24..31}.
static __device__ __forceinline__ v16bf load_b(const unsigned short* base, int ld, int lane) {
  int col = lane & 15, kh = lane >> 4;
  const unsigned short* p = base + (size_t)col * (size_t)ld + kh * 8;
  FragU f;
  f.u[0] = *(const uint4*)(p);
  f.u[1] = *(const uint4*)(p + 16);
  return f.v;
}

static __device__ __forceinline__ v8f vzero() {
  v8f z = {0.f, 0.f, 0.f, 0.f, 0.f, 0.f, 0.f, 0.f};
  return z;
}
static __device__ __forceinline__ v8f wmma_bf16(v16bf a, v16bf b, v8f c) {
  return __builtin_amdgcn_wmma_f32_16x16x32_bf16(false, a, false, b, (short)0, c, false, false);
}

// ---------------------------------------------------------------------------
// f32 -> bf16 pack
// ---------------------------------------------------------------------------
__global__ void pack_bf16_kernel(const float* __restrict__ src,
                                 unsigned short* __restrict__ dst, size_t n) {
  size_t stride = (size_t)gridDim.x * blockDim.x;
  for (size_t i = (size_t)blockIdx.x * blockDim.x + threadIdx.x; i < n; i += stride)
    dst[i] = f2bf(src[i]);
}

// ---------------------------------------------------------------------------
// C[m][n] = sum_k A[m][k] * W[n][k] + bias[n]   (i.e. A @ W^T + b)
// mode 0: store f32, 1: store silu(f32), 2: store bf16.
// 8 waves per block; each wave owns a 64x64 tile (4x4 grid of 16x16 WMMA).
// Block tile = 128 x 256.
// ---------------------------------------------------------------------------
__global__ __launch_bounds__(256) void gemm_bf16_nt(
    const unsigned short* __restrict__ A, const unsigned short* __restrict__ W,
    const float* __restrict__ bias, void* __restrict__ Cout,
    int M, int Nn, int K, int mode) {
  int lane = threadIdx.x & 31;
  int wave = threadIdx.x >> 5;
  int wm = blockIdx.x * 128 + (wave >> 2) * 64;
  int wn = blockIdx.y * 256 + (wave & 3) * 64;

  v8f acc[4][4];
#pragma unroll
  for (int i = 0; i < 4; ++i)
#pragma unroll
    for (int j = 0; j < 4; ++j) acc[i][j] = vzero();

  for (int k0 = 0; k0 < K; k0 += 32) {
    v16bf a[4], b[4];
#pragma unroll
    for (int i = 0; i < 4; ++i)
      a[i] = load_a(A + (size_t)(wm + 16 * i) * K + k0, K, lane);
#pragma unroll
    for (int j = 0; j < 4; ++j)
      b[j] = load_b(W + (size_t)(wn + 16 * j) * K + k0, K, lane);
    if (k0 + 32 < K) {   // hint next K-step into cache (global_prefetch)
      __builtin_prefetch(A + (size_t)wm * K + k0 + 32, 0, 1);
      __builtin_prefetch(W + (size_t)wn * K + k0 + 32, 0, 1);
    }
#pragma unroll
    for (int i = 0; i < 4; ++i)
#pragma unroll
      for (int j = 0; j < 4; ++j)
        acc[i][j] = wmma_bf16(a[i], b[j], acc[i][j]);
  }

  int kh = lane >> 4, nn = lane & 15;
#pragma unroll
  for (int i = 0; i < 4; ++i)
#pragma unroll
    for (int j = 0; j < 4; ++j) {
      int col = wn + 16 * j + nn;
      float bi = bias[col];
#pragma unroll
      for (int r = 0; r < 8; ++r) {
        int row = wm + 16 * i + r + kh * 8;   // C frag: VGPR r -> M=r / M=8+r
        float v = acc[i][j][r] + bi;
        size_t idx = (size_t)row * Nn + col;
        if (mode == 2) {
          ((unsigned short*)Cout)[idx] = f2bf(v);
        } else {
          if (mode == 1) v = v / (1.0f + __expf(-v));   // SiLU
          ((float*)Cout)[idx] = v;
        }
      }
    }
}

// ---------------------------------------------------------------------------
// Rotary (xPos interleaved pairs) on q and k; k also scaled 1/sqrt(D).
// fp32 in -> bf16 out. One thread per complex pair.
// ---------------------------------------------------------------------------
__global__ void rotary_kernel(const float* __restrict__ qf, const float* __restrict__ kf,
                              unsigned short* __restrict__ qb, unsigned short* __restrict__ kb,
                              const int* __restrict__ start_idx_p) {
  const int start = *start_idx_p;
  const float inv_sqrt_d = 0.08838834764831845f;   // 1/sqrt(128)
  const size_t total = (size_t)M_ * H_ * (D_ / 2);
  size_t stride = (size_t)gridDim.x * blockDim.x;
  for (size_t p = (size_t)blockIdx.x * blockDim.x + threadIdx.x; p < total; p += stride) {
    int pr = (int)(p & 63);              // pair index 0..63
    size_t rem = p >> 6;
    int h = (int)(rem % H_);
    size_t bn = rem / H_;
    int t = (int)(bn % N_);
    float theta = __powf(10000.0f, -(float)pr * (1.0f / 63.0f));
    float ang = (float)(start + t) * theta;
    float s, c;
    __sincosf(ang, &s, &c);
    size_t idx = bn * E_ + (size_t)h * 128 + (size_t)pr * 2;
    float a = qf[idx], b = qf[idx + 1];
    qb[idx]     = f2bf(a * c - b * s);
    qb[idx + 1] = f2bf(b * c + a * s);
    a = kf[idx]; b = kf[idx + 1];
    kb[idx]     = f2bf((a * c - b * s) * inv_sqrt_d);
    kb[idx + 1] = f2bf((b * c + a * s) * inv_sqrt_d);
  }
}

// ---------------------------------------------------------------------------
// Chunked retention scan. One workgroup (8 waves) per (b,h). State 128x128
// fp32 lives in registers: wave w owns d-rows [16w,16w+16) as 8 WMMA accs.
// kv_decay gamma^(63-j) is folded into a transposed bf16 copy of v (vdts),
// so  att2[i][j] = (q_i.k_j) * gamma^(i-63) * [i>=j]  gives
//   o  = att2 @ vdts^T + (q @ S) * gamma^(i+1)
//   S' = gamma^64 * S + k^T @ (decay*v)
// ---------------------------------------------------------------------------
__global__ __launch_bounds__(256) void retention_kernel(
    const unsigned short* __restrict__ qb, const unsigned short* __restrict__ kb,
    const unsigned short* __restrict__ vb, const float* __restrict__ prev_state,
    float* __restrict__ ret, float* __restrict__ state_out) {
  extern __shared__ unsigned short smem[];
  unsigned short* kts  = smem;                          // [128][64]  k^T
  unsigned short* vdts = smem + 128 * 64;               // [128][64]  (decay*v)^T
  unsigned short* atts = smem + 2 * 128 * 64;           // [64][64]   att2 bf16
  unsigned short* SbT  = smem + 2 * 128 * 64 + 64 * 64; // [128][128] S^T bf16

  const int bh = blockIdx.x;
  const int b = bh / H_, h = bh % H_;
  const int lane = threadIdx.x & 31, wave = threadIdx.x >> 5;
  const int kh = lane >> 4, nn = lane & 15;
  const float gamma = 1.0f - exp2f(-5.0f - (float)h);
  const float lg = __logf(gamma);
  const float g64 = __expf(64.0f * lg);
  const size_t head_off = (size_t)h * 128;

  // init register state from prev_state (B,H,D,D)
  v8f S[8];
  {
    const float* ps = prev_state + (size_t)(b * H_ + h) * D_ * D_;
#pragma unroll
    for (int eb = 0; eb < 8; ++eb)
#pragma unroll
      for (int r = 0; r < 8; ++r) {
        int d = 16 * wave + r + kh * 8;
        int e = 16 * eb + nn;
        S[eb][r] = ps[(size_t)d * D_ + e];
      }
  }

  for (int c = 0; c < NC_; ++c) {
    const size_t tok0 = (size_t)b * N_ + (size_t)c * C_;

    // (a) stage k^T, (decay*v)^T and bf16 snapshot of S^T into LDS
    for (int idx = threadIdx.x; idx < 128 * 64; idx += 256) {
      int d = idx >> 6, j = idx & 63;
      kts[idx] = kb[(tok0 + j) * E_ + head_off + d];
    }
    for (int idx = threadIdx.x; idx < 128 * 64; idx += 256) {
      int e = idx >> 6, j = idx & 63;
      float dec = __expf((float)(63 - j) * lg);
      vdts[idx] = f2bf(bf2f(vb[(tok0 + j) * E_ + head_off + e]) * dec);
    }
#pragma unroll
    for (int eb = 0; eb < 8; ++eb)
#pragma unroll
      for (int r = 0; r < 8; ++r) {
        int d = 16 * wave + r + kh * 8;
        int e = 16 * eb + nn;
        SbT[(size_t)e * 128 + d] = f2bf(S[eb][r]);
      }
    __syncthreads();

    // (b) att tiles: 16 tiles of 16x16 over (i,j), 2 per wave, K = 128
    for (int tt = wave * 2; tt < wave * 2 + 2; ++tt) {
      int ib = tt >> 2, jb = tt & 3;
      v8f acc = vzero();
#pragma unroll
      for (int k0 = 0; k0 < 128; k0 += 32) {
        v16bf a  = load_a(qb + (tok0 + 16 * ib) * E_ + head_off + k0, E_, lane);
        v16bf bf = load_b(kb + (tok0 + 16 * jb) * E_ + head_off + k0, E_, lane);
        acc = wmma_bf16(a, bf, acc);
      }
#pragma unroll
      for (int r = 0; r < 8; ++r) {
        int i = 16 * ib + r + kh * 8;
        int j = 16 * jb + nn;
        float v = (i >= j) ? acc[r] * __expf((float)(i - 63) * lg) : 0.0f;
        atts[(size_t)i * 64 + j] = f2bf(v);
      }
    }
    __syncthreads();

    // (c) output tiles: 32 tiles over (i-block 0..3, e-block 0..7), 4/wave
    for (int oi = 0; oi < 4; ++oi) {
      int tile = wave * 4 + oi;
      int ib = tile >> 3, eb = tile & 7;
      v8f a1 = vzero(), a2 = vzero();
#pragma unroll
      for (int k0 = 0; k0 < 64; k0 += 32) {   // att2 @ (decay*v)
        v16bf a  = load_a(atts + (size_t)(16 * ib) * 64 + k0, 64, lane);
        v16bf bf = load_b(vdts + (size_t)(16 * eb) * 64 + k0, 64, lane);
        a1 = wmma_bf16(a, bf, a1);
      }
#pragma unroll
      for (int k0 = 0; k0 < 128; k0 += 32) {  // q @ S
        v16bf a  = load_a(qb + (tok0 + 16 * ib) * E_ + head_off + k0, E_, lane);
        v16bf bf = load_b(SbT + (size_t)(16 * eb) * 128 + k0, 128, lane);
        a2 = wmma_bf16(a, bf, a2);
      }
#pragma unroll
      for (int r = 0; r < 8; ++r) {
        int i = 16 * ib + r + kh * 8;
        int e = 16 * eb + nn;
        float cd = __expf((float)(i + 1) * lg);   // cross_decay
        ret[(tok0 + i) * E_ + head_off + e] = a1[r] + a2[r] * cd;
      }
    }

    // state update: S = gamma^64 * S + k^T @ (decay*v); A frags shared per wave
    {
      v16bf a0 = load_a(kts + (size_t)(16 * wave) * 64 + 0,  64, lane);
      v16bf a1 = load_a(kts + (size_t)(16 * wave) * 64 + 32, 64, lane);
#pragma unroll
      for (int eb = 0; eb < 8; ++eb) {
        v8f s = S[eb] * g64;
        v16bf b0 = load_b(vdts + (size_t)(16 * eb) * 64 + 0,  64, lane);
        v16bf b1 = load_b(vdts + (size_t)(16 * eb) * 64 + 32, 64, lane);
        s = wmma_bf16(a0, b0, s);
        s = wmma_bf16(a1, b1, s);
        S[eb] = s;
      }
    }
    __syncthreads();
  }

  // final state out (B,H,D,D) fp32
  {
    float* so = state_out + (size_t)(b * H_ + h) * D_ * D_;
#pragma unroll
    for (int eb = 0; eb < 8; ++eb)
#pragma unroll
      for (int r = 0; r < 8; ++r) {
        int d = 16 * wave + r + kh * 8;
        int e = 16 * eb + nn;
        so[(size_t)d * D_ + e] = S[eb][r];
      }
  }
}

// ---------------------------------------------------------------------------
// GroupNorm(per head, affine=False) * silu-gate -> bf16. One wave per group
// of 128 (b,n,h); lane holds 4 values; wave32 shfl_xor reduction.
// ---------------------------------------------------------------------------
__global__ __launch_bounds__(256) void gnorm_gate_kernel(
    const float* __restrict__ ret, const float* __restrict__ gate,
    unsigned short* __restrict__ yb) {
  int lane = threadIdx.x & 31, wave = threadIdx.x >> 5;
  size_t g = (size_t)blockIdx.x * 8 + wave;   // group over B*N*H
  size_t bn = g / H_;
  int h = (int)(g % H_);
  size_t base = bn * E_ + (size_t)h * 128 + (size_t)lane * 4;
  float4 x = *(const float4*)(ret + base);
  float s  = x.x + x.y + x.z + x.w;
  float ss = x.x * x.x + x.y * x.y + x.z * x.z + x.w * x.w;
#pragma unroll
  for (int m = 16; m >= 1; m >>= 1) {
    s  += __shfl_xor(s, m, 32);
    ss += __shfl_xor(ss, m, 32);
  }
  float mu  = s * (1.0f / 128.0f);
  float var = ss * (1.0f / 128.0f) - mu * mu;
  float rs  = rsqrtf(var + 1e-6f);
  float4 gt = *(const float4*)(gate + base);   // silu already applied
  ushort4 o;
  o.x = f2bf((x.x - mu) * rs * gt.x);
  o.y = f2bf((x.y - mu) * rs * gt.y);
  o.z = f2bf((x.z - mu) * rs * gt.z);
  o.w = f2bf((x.w - mu) * rs * gt.w);
  *(ushort4*)(yb + base) = o;
}

// ---------------------------------------------------------------------------
extern "C" void kernel_launch(void* const* d_in, const int* in_sizes, int n_in,
                              void* d_out, int out_size, void* d_ws, size_t ws_size,
                              hipStream_t stream) {
  (void)in_sizes; (void)n_in; (void)out_size; (void)ws_size;
  const float* x   = (const float*)d_in[0];
  const float* ps  = (const float*)d_in[1];
  const float* Wq  = (const float*)d_in[2];  const float* bq = (const float*)d_in[3];
  const float* Wk  = (const float*)d_in[4];  const float* bk = (const float*)d_in[5];
  const float* Wv  = (const float*)d_in[6];  const float* bv = (const float*)d_in[7];
  const float* Wg  = (const float*)d_in[8];  const float* bg = (const float*)d_in[9];
  const float* Wo  = (const float*)d_in[10]; const float* bo = (const float*)d_in[11];
  const int* start_idx = (const int*)d_in[12];

  // workspace layout
  char* ws = (char*)d_ws;
  size_t o = 0;
  unsigned short* xb  = (unsigned short*)(ws + o); o += (size_t)M_ * E_ * 2;
  unsigned short* Wqb = (unsigned short*)(ws + o); o += (size_t)E_ * E_ * 2;
  unsigned short* Wkb = (unsigned short*)(ws + o); o += (size_t)E_ * E_ * 2;
  unsigned short* Wvb = (unsigned short*)(ws + o); o += (size_t)E_ * E_ * 2;
  unsigned short* Wgb = (unsigned short*)(ws + o); o += (size_t)E_ * E_ * 2;
  unsigned short* Wob = (unsigned short*)(ws + o); o += (size_t)E_ * E_ * 2;
  float* qf = (float*)(ws + o); o += (size_t)M_ * E_ * 4;   // reused as `ret`
  float* kf = (float*)(ws + o); o += (size_t)M_ * E_ * 4;   // reused as `gate`
  unsigned short* vbb = (unsigned short*)(ws + o); o += (size_t)M_ * E_ * 2;
  unsigned short* qbb = (unsigned short*)(ws + o); o += (size_t)M_ * E_ * 2;
  unsigned short* kbb = (unsigned short*)(ws + o); o += (size_t)M_ * E_ * 2;
  unsigned short* yb  = (unsigned short*)(ws + o); o += (size_t)M_ * E_ * 2;
  float* ret  = qf;   // qf consumed by rotary before retention writes ret
  float* gate = kf;   // kf consumed by rotary before gate GEMM writes here

  const int lds_bytes = (2 * 128 * 64 + 64 * 64 + 128 * 128) * 2;   // 73728
  hipFuncSetAttribute((const void*)retention_kernel,
                      hipFuncAttributeMaxDynamicSharedMemorySize, lds_bytes);

  // 1) pack fp32 -> bf16
  pack_bf16_kernel<<<2048, 256, 0, stream>>>(x,  xb,  (size_t)M_ * E_);
  pack_bf16_kernel<<<2048, 256, 0, stream>>>(Wq, Wqb, (size_t)E_ * E_);
  pack_bf16_kernel<<<2048, 256, 0, stream>>>(Wk, Wkb, (size_t)E_ * E_);
  pack_bf16_kernel<<<2048, 256, 0, stream>>>(Wv, Wvb, (size_t)E_ * E_);
  pack_bf16_kernel<<<2048, 256, 0, stream>>>(Wg, Wgb, (size_t)E_ * E_);
  pack_bf16_kernel<<<2048, 256, 0, stream>>>(Wo, Wob, (size_t)E_ * E_);

  dim3 gg(M_ / 128, E_ / 256);   // 32 x 8 blocks
  // 2) projections
  gemm_bf16_nt<<<gg, 256, 0, stream>>>(xb, Wqb, bq, qf,  M_, E_, E_, 0);
  gemm_bf16_nt<<<gg, 256, 0, stream>>>(xb, Wkb, bk, kf,  M_, E_, E_, 0);
  // 3) rotary + scale -> bf16 q,k
  rotary_kernel<<<4096, 256, 0, stream>>>(qf, kf, qbb, kbb, start_idx);
  // 4) v (bf16) and gate (silu fp32, overwrites kf after rotary consumed it)
  gemm_bf16_nt<<<gg, 256, 0, stream>>>(xb, Wvb, bv, vbb,  M_, E_, E_, 2);
  gemm_bf16_nt<<<gg, 256, 0, stream>>>(xb, Wgb, bg, gate, M_, E_, E_, 1);
  // 5) retention scan -> ret (fp32) + final state into d_out tail
  float* state_out = (float*)d_out + (size_t)M_ * E_;
  retention_kernel<<<B_ * H_, 256, lds_bytes, stream>>>(qbb, kbb, vbb, ps, ret, state_out);
  // 6) groupnorm * gate -> bf16
  gnorm_gate_kernel<<<(M_ * H_) / 8, 256, 0, stream>>>(ret, gate, yb);
  // 7) output projection -> fp32 out
  gemm_bf16_nt<<<gg, 256, 0, stream>>>(yb, Wob, bo, (float*)d_out, M_, E_, E_, 0);
}
